// Model_4518305595855
// MI455X (gfx1250) — matrix operand order repaired and checked by reading, not compiled, not run on hardware
//
#include <hip/hip_runtime.h>
#include <hip/hip_bf16.h>

// ---------------------------------------------------------------------------
// Momentum-LSTM (SEQ=784, B=128, H=256) as a persistent WMMA kernel.
//   grid = 16 workgroups x 256 threads (8 waves).  Workgroup jt owns h-columns
//   [16*jt, 16*jt+16) and keeps its 64x256 f16 slice of W_hh in LDS for the
//   whole scan.  Wave w owns batch rows [16*w, 16*w+16).  c and v state stay
//   in VGPRs; h ping-pongs through L2 with a device-wide barrier per step.
// ---------------------------------------------------------------------------

#define MU   0.6f
#define EPSC 0.6f
#define HID  256
#define SEQ_ 784
#define BATCH_ 128

typedef _Float16 v8h  __attribute__((ext_vector_type(8)));
typedef _Float16 v16h __attribute__((ext_vector_type(16)));
typedef float    v8f  __attribute__((ext_vector_type(8)));

union Frag16 { v16h v; v8h h[2]; };

// ---- workspace layout (bytes) ---------------------------------------------
#define OFF_BAR  ((size_t)0)
#define OFF_H0   ((size_t)256)
#define OFF_H1   (OFF_H0 + (size_t)BATCH_ * HID * 2)
#define OFF_XT   (OFF_H1 + (size_t)BATCH_ * HID * 2)
#define OFF_HF   (OFF_XT + (size_t)SEQ_ * BATCH_ * 4)
// total need = OFF_HF + 128*256*4  ~= 0.64 MB

// ---------------------------------------------------------------------------
__global__ void k_init(const float* __restrict__ x, const int* __restrict__ perm,
                       float* __restrict__ xt, _Float16* __restrict__ h0,
                       unsigned* __restrict__ bar) {
  int i = blockIdx.x * blockDim.x + threadIdx.x;
  if (i < 2) bar[i] = 0u;
  if (i < BATCH_ * HID) h0[i] = (_Float16)0.0f;
  if (i < SEQ_ * BATCH_) {
    int t = i >> 7;          // i / 128
    int b = i & 127;
    xt[i] = x[b * SEQ_ + perm[t]];   // xs[t][b] = x[b][perm[t]]
  }
}

// ---------------------------------------------------------------------------
__device__ __forceinline__ void grid_barrier(unsigned* bar, unsigned nwg) {
  __threadfence();                       // release our h stores to device scope
  __syncthreads();
  if (threadIdx.x == 0) {
    unsigned gen = __hip_atomic_load(bar + 1, __ATOMIC_RELAXED, __HIP_MEMORY_SCOPE_AGENT);
    unsigned prev = __hip_atomic_fetch_add(bar, 1u, __ATOMIC_ACQ_REL, __HIP_MEMORY_SCOPE_AGENT);
    if (prev == nwg - 1u) {
      __hip_atomic_store(bar, 0u, __ATOMIC_RELAXED, __HIP_MEMORY_SCOPE_AGENT);
      __hip_atomic_fetch_add(bar + 1, 1u, __ATOMIC_RELEASE, __HIP_MEMORY_SCOPE_AGENT);
    } else {
      while (__hip_atomic_load(bar + 1, __ATOMIC_ACQUIRE, __HIP_MEMORY_SCOPE_AGENT) == gen)
        __builtin_amdgcn_s_sleep(2);
    }
  }
  __syncthreads();
  __threadfence();                       // acquire: drop stale WGP$ lines
}

__device__ __forceinline__ float sigmf(float x) {
  return 1.0f / (1.0f + __expf(-x));
}

// ---------------------------------------------------------------------------
__launch_bounds__(256)
__global__ void k_lstm(const float* __restrict__ W_hh,  // (1024,256) row-major
                       const float* __restrict__ W_ih,  // (1024,)
                       const float* __restrict__ b_ih,  // (1024,)
                       const float* __restrict__ b_hh,  // (1024,)
                       const float* __restrict__ xt,    // (784,128)
                       _Float16* __restrict__ h0, _Float16* __restrict__ h1,
                       float* __restrict__ hf32,        // (128,256) final h
                       unsigned* __restrict__ bar) {
  // 64 rows (4 gates x 16 cols) x 256 K, padded to 264 halves -> 33792 B LDS
  __shared__ _Float16 lds_w[64 * 264];

  const int jt   = blockIdx.x;        // h-column tile 0..15
  const int tid  = threadIdx.x;
  const int wave = tid >> 5;          // 0..7  -> batch tile
  const int lane = tid & 31;
  const int nl   = lane & 15;         // column within tile (WMMA C: N = lane%16)
  const int half = lane >> 4;         // WMMA C: M = r + 8*half
  const int m0   = wave * 16;
  const int mbase = m0 + half * 8;
  const int hcol  = jt * 16 + nl;

  // --- stage this WG's W_hh slice into LDS as f16 (once, stays all 784 steps)
  for (int idx = tid; idx < 64 * 256; idx += 256) {
    int row = idx >> 8;               // g*16 + rr
    int k   = idx & 255;
    int g   = row >> 4;
    int rr  = row & 15;
    int n   = g * 256 + jt * 16 + rr;
    lds_w[row * 264 + k] = (_Float16)W_hh[n * 256 + k];
  }
  __syncthreads();

  // --- per-lane gate constants (one n-column per gate per lane)
  float w_ih_c[4], b_ih_c[4], b_hh_c[4];
#pragma unroll
  for (int g = 0; g < 4; ++g) {
    int n = g * 256 + jt * 16 + nl;
    w_ih_c[g] = W_ih[n];
    b_ih_c[g] = b_ih[n];
    b_hh_c[g] = b_hh[n];
  }

  // --- register-resident state: c tile + momentum v for 4 gate tiles
  v8f cfrag = {};
  v8f vfrag[4];
#pragma unroll
  for (int g = 0; g < 4; ++g) vfrag[g] = (v8f){};

  const int mA = m0 + nl;             // A-fragment row (A: M = lane%16)
  const int kb = half * 8;            // A K-group base per ISA layout

  for (int t = 0; t < SEQ_; ++t) {
    const _Float16* __restrict__ hsrc = (t & 1) ? h1 : h0;
    _Float16* __restrict__       hdst = (t & 1) ? h0 : h1;

    // ---- A fragments: h[mA][k] per ISA 16-bit A 16x32 layout ----
    Frag16 a[8];
    const _Float16* hr = hsrc + (size_t)mA * HID;
#pragma unroll
    for (int kk = 0; kk < 8; ++kk) {
      a[kk].h[0] = *(const v8h*)(hr + kk * 32 + kb);
      a[kk].h[1] = *(const v8h*)(hr + kk * 32 + kb + 16);
    }

    if (t + 1 < SEQ_)
      __builtin_prefetch(xt + (size_t)(t + 1) * BATCH_ + mbase, 0, 3);

    // ---- 4 gate tiles, K = 256 -> 8 WMMAs each (B streamed from LDS) ----
    v8f acc[4];
#pragma unroll
    for (int g = 0; g < 4; ++g) {
      v8f c = {};
      const _Float16* wr = &lds_w[(g * 16 + nl) * 264 + half * 8];
#pragma unroll
      for (int kk = 0; kk < 8; ++kk) {
        Frag16 bfr;
        bfr.h[0] = *(const v8h*)(wr + kk * 32);
        bfr.h[1] = *(const v8h*)(wr + kk * 32 + 16);
        c = __builtin_amdgcn_wmma_f32_16x16x32_f16(
                /*neg_a=*/false, a[kk].v, /*neg_b=*/false, bfr.v,
                /*c_mod=*/(short)0, c, /*reuse_a=*/false, /*reuse_b=*/false);
      }
      acc[g] = c;
    }

    // ---- elementwise LSTM update (c, v in registers) ----
    float xv[8];
    const float* xrow = xt + (size_t)t * BATCH_ + mbase;
#pragma unroll
    for (int r = 0; r < 8; ++r) xv[r] = xrow[r];

#pragma unroll
    for (int r = 0; r < 8; ++r) {
      float gt[4];
#pragma unroll
      for (int g = 0; g < 4; ++g) {
        float pre = xv[r] * w_ih_c[g] + b_ih_c[g];
        float vv  = MU * vfrag[g][r] + EPSC * pre;
        vfrag[g][r] = vv;
        gt[g] = vv + acc[g][r] + b_hh_c[g];
      }
      float ig = sigmf(gt[0]);
      float fg = sigmf(gt[1]);
      float gg = tanhf(gt[2]);
      float og = sigmf(gt[3]);
      float cc = fg * cfrag[r] + ig * gg;
      cfrag[r] = cc;
      float hv = og * tanhf(cc);
      hdst[(size_t)(mbase + r) * HID + hcol] = (_Float16)hv;
      if (t == SEQ_ - 1) hf32[(size_t)(mbase + r) * HID + hcol] = hv;
    }

    grid_barrier(bar, gridDim.x);
  }
}

// ---------------------------------------------------------------------------
__global__ void k_head(const float* __restrict__ hf32,
                       const float* __restrict__ W_lin,  // (10,256)
                       const float* __restrict__ b_lin,  // (10,)
                       float* __restrict__ out) {        // (128,10)
  int idx = blockIdx.x * blockDim.x + threadIdx.x;
  if (idx >= BATCH_ * 10) return;
  int b = idx / 10, j = idx % 10;
  float s = b_lin[j];
  const float* hr = hf32 + (size_t)b * HID;
  const float* wr = W_lin + (size_t)j * HID;
#pragma unroll 8
  for (int k = 0; k < HID; ++k) s += hr[k] * wr[k];
  out[idx] = s;
}

// ---------------------------------------------------------------------------
extern "C" void kernel_launch(void* const* d_in, const int* in_sizes, int n_in,
                              void* d_out, int out_size, void* d_ws, size_t ws_size,
                              hipStream_t stream) {
  const float* x     = (const float*)d_in[0];   // (128,784)
  const int*   perm  = (const int*)  d_in[1];   // (784,)
  const float* W_ih  = (const float*)d_in[2];   // (1024,1)
  const float* b_ih  = (const float*)d_in[3];   // (1024,)
  const float* W_hh  = (const float*)d_in[4];   // (1024,256)
  const float* b_hh  = (const float*)d_in[5];   // (1024,)
  const float* W_lin = (const float*)d_in[6];   // (10,256)
  const float* b_lin = (const float*)d_in[7];   // (10,)
  float* out = (float*)d_out;

  char* ws = (char*)d_ws;
  unsigned* bar = (unsigned*)(ws + OFF_BAR);
  _Float16* h0  = (_Float16*)(ws + OFF_H0);
  _Float16* h1  = (_Float16*)(ws + OFF_H1);
  float*    xt  = (float*)   (ws + OFF_XT);
  float*    hf  = (float*)   (ws + OFF_HF);

  // prep: permuted input, zeroed h(0), zeroed barrier (re-done every launch)
  k_init<<<(SEQ_ * BATCH_ + 255) / 256, 256, 0, stream>>>(x, perm, xt, h0, bar);
  // persistent scan: 16 WGs x 8 waves = 128 wave-units = full per-step GEMM
  k_lstm<<<16, 256, 0, stream>>>(W_hh, W_ih, b_ih, b_hh, xt, h0, h1, hf, bar);
  // tiny head: (128,256) @ (256,10)
  k_head<<<(BATCH_ * 10 + 255) / 256, 256, 0, stream>>>(hf, W_lin, b_lin, out);
}
